// SwinTransformerBlock_37349035606804
// MI455X (gfx1250) — matrix operand order repaired and checked by reading, not compiled, not run on hardware
//
#include <hip/hip_runtime.h>
#include <hip/hip_bf16.h>
#include <hip/hip_fp16.h>

typedef _Float16 hh;
typedef _Float16 v16h __attribute__((ext_vector_type(16)));
typedef float    v8f  __attribute__((ext_vector_type(8)));

union AF { uint4 u4[2]; v16h v; };
union U8 { uint4 u; hh h[8]; };

#define LTOT   21952        // 7*56*56
#define MROWS  22016        // 64 windows * 344 tokens
#define NTOK   344
#define NPADT  352

// ---------------------------------------------------------------------------
// fp32 -> fp16 weight conversion
// ---------------------------------------------------------------------------
__global__ void f32_to_f16(const float* __restrict__ src, hh* __restrict__ dst, int n) {
  int i = blockIdx.x * 256 + threadIdx.x;
  if (i < n) dst[i] = (hh)src[i];
}

// ---------------------------------------------------------------------------
// Precompute relative-position bias table per head: biasP[h][n][m] (f16)
// ---------------------------------------------------------------------------
__global__ void build_bias(const float* __restrict__ bt, hh* __restrict__ biasP) {
  int id = blockIdx.x * 256 + threadIdx.x;
  if (id >= 12 * NTOK * NTOK) return;
  const int h = id / (NTOK * NTOK);
  const int rem = id - h * (NTOK * NTOK);
  const int n = rem / NTOK, m = rem % NTOK;
  float v = 0.f;
  if (n > 0 && m > 0) {
    const int i = n - 1, j = m - 1;
    const int zi = i / 49, yi = (i / 7) % 7, xi = i % 7;
    const int zj = j / 49, yj = (j / 7) % 7, xj = j % 7;
    const int idx = (zi - zj + 6) * 20 + (yi - yj + 6) * 13 + (xi - xj + 6);
    v = bt[idx * 12 + h];
  }
  biasP[id] = (hh)v;
}

// ---------------------------------------------------------------------------
// Precompute shifted-window region id per (window, token): regT[64][344] (u8)
// ---------------------------------------------------------------------------
__global__ void build_reg(unsigned char* __restrict__ regT) {
  int id = blockIdx.x * 256 + threadIdx.x;
  if (id >= 64 * NTOK) return;
  const int win = id / NTOK, t = id % NTOK;
  unsigned char v = 0;
  if (t > 0) {
    const int i = t - 1;
    const int zi = i / 49, yi = (i / 7) % 7, xi = i % 7;
    const int ih = win >> 3, iw = win & 7;
    const int gy = ih * 7 + yi, gx = iw * 7 + xi;
    v = (unsigned char)((zi < 4 ? 1 : 2) * 9
                      + (gy < 49 ? 0 : (gy < 53 ? 1 : 2)) * 3
                      + (gx < 49 ? 0 : (gx < 53 ? 1 : 2)));
  }
  regT[id] = v;
}

// ---------------------------------------------------------------------------
// LN1 + roll(-3) + window partition + gt concat  ->  xc f16 [22016][384]
// ---------------------------------------------------------------------------
__global__ __launch_bounds__(128) void ln1_gather(
    const float* __restrict__ x, const float* __restrict__ gt,
    const float* __restrict__ g, const float* __restrict__ b,
    hh* __restrict__ xc)
{
  const int r = blockIdx.x;            // xc row
  const int tid = threadIdx.x;
  const int win = r / NTOK, t = r % NTOK;
  if (t == 0) {
    for (int c = tid; c < 384; c += 128)
      xc[(size_t)r * 384 + c] = (hh)gt[(size_t)win * 384 + c];
    return;
  }
  const int i  = t - 1;
  const int zi = i / 49, yi = (i / 7) % 7, xi = i % 7;
  const int ih = win >> 3, iw = win & 7;
  const int oz = (zi + 3) % 7;
  const int oy = (ih * 7 + yi + 3) % 56;
  const int ox = (iw * 7 + xi + 3) % 56;
  const size_t L = ((size_t)oz * 56 + oy) * 56 + ox;
  const float* xr = x + L * 384;

  __shared__ float red[128];
  float s = 0.f;
  for (int c = tid; c < 384; c += 128) s += xr[c];
  red[tid] = s; __syncthreads();
  for (int o = 64; o > 0; o >>= 1) { if (tid < o) red[tid] += red[tid + o]; __syncthreads(); }
  float mu = red[0] * (1.0f / 384.0f);
  __syncthreads();
  float v = 0.f;
  for (int c = tid; c < 384; c += 128) { float d = xr[c] - mu; v += d * d; }
  red[tid] = v; __syncthreads();
  for (int o = 64; o > 0; o >>= 1) { if (tid < o) red[tid] += red[tid + o]; __syncthreads(); }
  float rs = rsqrtf(red[0] * (1.0f / 384.0f) + 1e-5f);
  for (int c = tid; c < 384; c += 128)
    xc[(size_t)r * 384 + c] = (hh)((xr[c] - mu) * rs * g[c] + b[c]);
}

// ---------------------------------------------------------------------------
// LN2:  xres f32 [21952][384] -> ln2 f16
// ---------------------------------------------------------------------------
__global__ __launch_bounds__(128) void ln2_kernel(
    const float* __restrict__ xres, const float* __restrict__ g,
    const float* __restrict__ b, hh* __restrict__ outp)
{
  const int r = blockIdx.x;
  const int tid = threadIdx.x;
  const float* xr = xres + (size_t)r * 384;
  __shared__ float red[128];
  float s = 0.f;
  for (int c = tid; c < 384; c += 128) s += xr[c];
  red[tid] = s; __syncthreads();
  for (int o = 64; o > 0; o >>= 1) { if (tid < o) red[tid] += red[tid + o]; __syncthreads(); }
  float mu = red[0] * (1.0f / 384.0f);
  __syncthreads();
  float v = 0.f;
  for (int c = tid; c < 384; c += 128) { float d = xr[c] - mu; v += d * d; }
  red[tid] = v; __syncthreads();
  for (int o = 64; o > 0; o >>= 1) { if (tid < o) red[tid] += red[tid + o]; __syncthreads(); }
  float rs = rsqrtf(red[0] * (1.0f / 384.0f) + 1e-5f);
  for (int c = tid; c < 384; c += 128)
    outp[(size_t)r * 384 + c] = (hh)((xr[c] - mu) * rs * g[c] + b[c]);
}

// ---------------------------------------------------------------------------
// Windowed attention: one workgroup per (window, head, 64-row q panel).
// S (f32) is pre-initialized with bias+mask and then used directly as the
// WMMA C-operand (load C fragment from LDS, wmma, store back) — no separate
// accumulate pass, no WMMA->VALU hazard nops.
// ---------------------------------------------------------------------------
#define SM_P_OFF  (64 * 352 * 4)
#define SM_VT_OFF (64 * 352 * 4 + 64 * 352 * 2)
#define SMEM_ATTN (64 * 352 * 4 + 64 * 352 * 2 + 32 * 360 * 2)

__global__ __launch_bounds__(256) void attn_win(
    const hh* __restrict__ qh, const hh* __restrict__ kh, const hh* __restrict__ vh,
    const hh* __restrict__ biasP, const unsigned char* __restrict__ regT,
    hh* __restrict__ attnbuf)
{
  extern __shared__ char smem[];
  float* S  = (float*)smem;                  // [64][352]
  hh*    P  = (hh*)(smem + SM_P_OFF);        // [64][352]
  hh*    VT = (hh*)(smem + SM_VT_OFF);       // [32][360]

  const int win = blockIdx.x, head = blockIdx.y, panel = blockIdx.z;
  const int row0 = panel * 64;
  const size_t base = ((size_t)(win * 12 + head)) * NPADT * 32;
  const hh* Q  = qh + base;
  const hh* Km = kh + base;
  const hh* V  = vh + base;
  const hh* bh = biasP + (size_t)head * NTOK * NTOK;
  const unsigned char* rw = regT + win * NTOK;

  const int tid  = threadIdx.x;
  const int wave = tid >> 5, lane = tid & 31;
  const int half = lane >> 4, ln = lane & 15;

  // preload V transposed into LDS
  for (int i = tid; i < NPADT * 32; i += 256) {
    int k = i >> 5, dd = i & 31;
    VT[dd * 360 + k] = V[i];
  }

  // init S with bias + mask (branchless; tables padded so loads are safe)
  {
    const int row = tid >> 2, part = tid & 3;
    const int n = row0 + row;
    const bool nvalid = (n < NTOK);
    const int nc = nvalid ? n : 0;
    const unsigned char rn = rw[nc];
    const hh* brow = bh + (size_t)nc * NTOK;
    float* Srow = S + row * 352;
    for (int m = part; m < 352; m += 4) {
      float v = (float)brow[m];
      const unsigned char rm2 = rw[m];
      v = ((n != 0) & (m != 0) & (rn != rm2)) ? (v - 100.0f) : v;
      const bool ok = nvalid & (m < NTOK);
      Srow[m] = ok ? v : -1e30f;
    }
  }
  __syncthreads();

  // phase 1: S = (scaled Q) K^T + S   (wave owns one 16-row block, 11 key tiles)
  {
    const int rt = wave & 3;                 // q row tile within panel
    const int r16 = row0 + rt * 16;
    if (r16 < NPADT) {
      const int qrow = r16 + ln;
      AF a;
      a.u4[0] = *(const uint4*)(Q + qrow * 32 + half * 8);
      a.u4[1] = *(const uint4*)(Q + qrow * 32 + 16 + half * 8);
      for (int ct = (wave >> 2); ct < 22; ct += 2) {
        const int kcol = ct * 16 + ln;
        AF b;
        b.u4[0] = *(const uint4*)(Km + kcol * 32 + half * 8);
        b.u4[1] = *(const uint4*)(Km + kcol * 32 + 16 + half * 8);
        float* Scol = S + (rt * 16 + 8 * half) * 352 + ct * 16 + ln;
        v8f c;
#pragma unroll
        for (int r = 0; r < 8; ++r) c[r] = Scol[r * 352];       // bias+mask as C
        c = __builtin_amdgcn_wmma_f32_16x16x32_f16(false, a.v, false, b.v, (short)0, c, false, false);
#pragma unroll
        for (int r = 0; r < 8; ++r) Scol[r * 352] = c[r];
      }
    }
  }
  __syncthreads();

  // phase 2: row softmax, 4 threads per row, write P (f16)
  {
    const int row = tid >> 2, part = tid & 3;
    float* Srow = S + row * 352;
    float mx = -1e30f;
    for (int c = part; c < 352; c += 4) mx = fmaxf(mx, Srow[c]);
    mx = fmaxf(mx, __shfl_xor(mx, 1));
    mx = fmaxf(mx, __shfl_xor(mx, 2));
    float sum = 0.f;
    for (int c = part; c < 352; c += 4) sum += __expf(Srow[c] - mx);
    sum += __shfl_xor(sum, 1);
    sum += __shfl_xor(sum, 2);
    const float inv = 1.0f / sum;
    const bool valid = (row0 + row) < NTOK;
    hh* Prow = P + row * 352;
    for (int c = part; c < 352; c += 4)
      Prow[c] = valid ? (hh)(__expf(Srow[c] - mx) * inv) : (hh)0.0f;
  }
  __syncthreads();

  // phase 3: O = P V  (8 waves -> 4 row-tiles x 2 d-tiles)
  {
    const int rt = wave >> 1, cn = wave & 1;
    const int r16 = row0 + rt * 16;
    if (r16 < NPADT) {
      v8f c = {};
      const int lrow = rt * 16 + ln;
      const int vcol = cn * 16 + ln;
      for (int kt = 0; kt < 11; ++kt) {
        AF a, b;
        a.u4[0] = *(const uint4*)(&P[lrow * 352 + kt * 32 + half * 8]);
        a.u4[1] = *(const uint4*)(&P[lrow * 352 + kt * 32 + 16 + half * 8]);
        b.u4[0] = *(const uint4*)(&VT[vcol * 360 + kt * 32 + half * 8]);
        b.u4[1] = *(const uint4*)(&VT[vcol * 360 + kt * 32 + 16 + half * 8]);
        c = __builtin_amdgcn_wmma_f32_16x16x32_f16(false, a.v, false, b.v, (short)0, c, false, false);
      }
#pragma unroll
      for (int r = 0; r < 8; ++r) {
        const int tok = r16 + r + 8 * half;
        if (tok < NTOK)
          attnbuf[((size_t)(win * NTOK + tok)) * 384 + head * 32 + cn * 16 + ln] = (hh)c[r];
      }
    }
  }
}

// ---------------------------------------------------------------------------
// f16 WMMA GEMM, fully specialized per call site:
//   template<MODE, M, K, N>, 256 threads (8 waves as 4x2), 128x128 tile,
//   wave = 32x64, double-buffered LDS, 8 WMMAs per K-step (K-step = 32).
// MODE 0: qkv scatter (+bias, q scaled)  MODE 1: proj (+bias, unwindow+residual)
// MODE 2: fc1 (+bias, GELU, f16 out)     MODE 3: fc2 (+bias, +residual, f32 out)
// ---------------------------------------------------------------------------
template<int MODE, int M, int Kd, int N>
__global__ __launch_bounds__(256) void gemm_f16(
    const hh* __restrict__ A, const hh* __restrict__ B,
    const float* __restrict__ bias,
    float* __restrict__ outf,
    hh* __restrict__ qh, hh* __restrict__ kh, hh* __restrict__ vh,
    const float* __restrict__ xin,
    float* __restrict__ xres,
    hh* __restrict__ outh)
{
  __shared__ hh As[2][128][40];    // 128 rows x 32 k (+8 pad), double buffered
  __shared__ hh BsT[2][128][40];   // 128 cols x 32 k (+8 pad), double buffered

  const int tid  = threadIdx.x;
  const int wave = tid >> 5, lane = tid & 31;
  const int wm = wave >> 1, wn = wave & 1;
  const int half = lane >> 4, ln = lane & 15;
  const int rowbase = blockIdx.x * 128;
  const int colbase = blockIdx.y * 128;

  v8f acc[2][4] = {};

  auto stage = [&](int k0, int buf) {
#pragma unroll
    for (int i = 0; i < 2; ++i) {            // A tile: 512 uint4
      const int idx = tid + i * 256;
      const int r  = idx >> 2;
      const int cc = (idx & 3) * 8;
      int gr = rowbase + r; if (gr >= M) gr = M - 1;
      *(uint4*)(&As[buf][r][cc]) = *(const uint4*)(A + (size_t)gr * Kd + k0 + cc);
    }
#pragma unroll
    for (int i = 0; i < 2; ++i) {            // B tile 32x128 -> transposed
      const int idx = tid + i * 256;
      const int r  = idx >> 4;               // k row 0..31
      const int cc = (idx & 15) * 8;         // col 0..120
      U8 t; t.u = *(const uint4*)(B + (size_t)(k0 + r) * N + colbase + cc);
#pragma unroll
      for (int j = 0; j < 8; ++j) BsT[buf][cc + j][r] = t.h[j];
    }
  };

  stage(0, 0);
  __syncthreads();
  constexpr int nk = Kd >> 5;
  for (int ki = 0; ki < nk; ++ki) {
    const int buf = ki & 1;
    if (ki + 1 < nk) {
      __builtin_prefetch(A + (size_t)rowbase * Kd + ((ki + 1) << 5), 0, 1);
      stage((ki + 1) << 5, buf ^ 1);
    }
    AF a0, a1;
    const int ar0 = wm * 32 + ln, ar1 = ar0 + 16;
    a0.u4[0] = *(const uint4*)(&As[buf][ar0][half * 8]);
    a0.u4[1] = *(const uint4*)(&As[buf][ar0][16 + half * 8]);
    a1.u4[0] = *(const uint4*)(&As[buf][ar1][half * 8]);
    a1.u4[1] = *(const uint4*)(&As[buf][ar1][16 + half * 8]);
#pragma unroll
    for (int ni = 0; ni < 4; ++ni) {
      AF b;
      const int bc = wn * 64 + ni * 16 + ln;
      b.u4[0] = *(const uint4*)(&BsT[buf][bc][half * 8]);
      b.u4[1] = *(const uint4*)(&BsT[buf][bc][16 + half * 8]);
      acc[0][ni] = __builtin_amdgcn_wmma_f32_16x16x32_f16(false, a0.v, false, b.v, (short)0, acc[0][ni], false, false);
      acc[1][ni] = __builtin_amdgcn_wmma_f32_16x16x32_f16(false, a1.v, false, b.v, (short)0, acc[1][ni], false, false);
    }
    __syncthreads();
  }

  // epilogue (single mode per instantiation)
#pragma unroll
  for (int mi = 0; mi < 2; ++mi) {
#pragma unroll
    for (int ni = 0; ni < 4; ++ni) {
#pragma unroll
      for (int r = 0; r < 8; ++r) {
        const int grow = rowbase + wm * 32 + mi * 16 + r + 8 * half;
        const int gcol = colbase + wn * 64 + ni * 16 + ln;
        if (grow >= M) continue;
        float val = acc[mi][ni][r] + bias[gcol];
        if constexpr (MODE == 0) {
          const int which = (gcol >= 768) ? 2 : ((gcol >= 384) ? 1 : 0);
          const int rem = gcol - which * 384;
          const int hd = rem >> 5, dd = rem & 31;
          const int win = grow / NTOK, tok = grow - win * NTOK;
          if (which == 0) val *= 0.17677669529663687f;     // d^-0.5
          hh* dst = (which == 0) ? qh : ((which == 1) ? kh : vh);
          dst[(((size_t)(win * 12 + hd)) * NPADT + tok) * 32 + dd] = (hh)val;
        } else if constexpr (MODE == 1) {
          const int win = grow / NTOK, tok = grow - win * NTOK;
          if (tok == 0) {
            outf[(size_t)LTOT * 384 + (size_t)win * 384 + gcol] = val;  // gt_out
          } else {
            const int i = tok - 1;
            const int zi = i / 49, yi = (i / 7) % 7, xi = i % 7;
            const int ih = win >> 3, iw = win & 7;
            const int oz = (zi + 3) % 7;
            const int oy = (ih * 7 + yi + 3) % 56;
            const int ox = (iw * 7 + xi + 3) % 56;
            const size_t L = ((size_t)oz * 56 + oy) * 56 + ox;
            xres[L * 384 + gcol] = xin[L * 384 + gcol] + val;
          }
        } else if constexpr (MODE == 2) {
          val = 0.5f * val * (1.0f + erff(val * 0.70710678118654752f));
          outh[(size_t)grow * N + gcol] = (hh)val;
        } else {
          outf[(size_t)grow * 384 + gcol] = xres[(size_t)grow * 384 + gcol] + val;
        }
      }
    }
  }
}

// ---------------------------------------------------------------------------
extern "C" void kernel_launch(void* const* d_in, const int* in_sizes, int n_in,
                              void* d_out, int out_size, void* d_ws, size_t ws_size,
                              hipStream_t stream)
{
  (void)in_sizes; (void)n_in; (void)out_size; (void)ws_size;
  const float* x          = (const float*)d_in[0];
  const float* gt         = (const float*)d_in[1];
  const float* norm1_g    = (const float*)d_in[2];
  const float* norm1_b    = (const float*)d_in[3];
  const float* qkv_w      = (const float*)d_in[4];
  const float* qkv_b      = (const float*)d_in[5];
  const float* bias_table = (const float*)d_in[6];
  const float* proj_w     = (const float*)d_in[7];
  const float* proj_b     = (const float*)d_in[8];
  const float* norm2_g    = (const float*)d_in[9];
  const float* norm2_b    = (const float*)d_in[10];
  const float* fc1_w      = (const float*)d_in[11];
  const float* fc1_b      = (const float*)d_in[12];
  const float* fc2_w      = (const float*)d_in[13];
  const float* fc2_b      = (const float*)d_in[14];
  float* out = (float*)d_out;

  char* ws = (char*)d_ws;
  size_t off = 0;
  auto alloc = [&](size_t bytes) -> void* {
    void* p = ws + off;
    off += (bytes + 255) & ~(size_t)255;
    return p;
  };
  hh* wqkv    = (hh*)alloc((size_t)384 * 1152 * 2);
  hh* wproj   = (hh*)alloc((size_t)384 * 384 * 2);
  hh* wfc1    = (hh*)alloc((size_t)384 * 1536 * 2);
  hh* wfc2    = (hh*)alloc((size_t)1536 * 384 * 2);
  hh* xc      = (hh*)alloc((size_t)MROWS * 384 * 2);
  hh* qh      = (hh*)alloc((size_t)64 * 12 * NPADT * 32 * 2);
  hh* kh      = (hh*)alloc((size_t)64 * 12 * NPADT * 32 * 2);
  hh* vh      = (hh*)alloc((size_t)64 * 12 * NPADT * 32 * 2);
  hh* attnbuf = (hh*)alloc((size_t)MROWS * 384 * 2);
  float* xres = (float*)alloc((size_t)LTOT * 384 * 4);
  hh* ln2h    = (hh*)alloc((size_t)LTOT * 384 * 2);
  hh* h1      = (hh*)alloc((size_t)LTOT * 1536 * 2);
  hh* biasP   = (hh*)alloc(((size_t)12 * NTOK * NTOK + 16) * 2);     // +pad
  unsigned char* regT = (unsigned char*)alloc((size_t)64 * NTOK + 16); // +pad

  // 1) weights to f16 + attention tables
  f32_to_f16<<<(384 * 1152 + 255) / 256, 256, 0, stream>>>(qkv_w, wqkv, 384 * 1152);
  f32_to_f16<<<(384 * 384 + 255) / 256, 256, 0, stream>>>(proj_w, wproj, 384 * 384);
  f32_to_f16<<<(384 * 1536 + 255) / 256, 256, 0, stream>>>(fc1_w, wfc1, 384 * 1536);
  f32_to_f16<<<(1536 * 384 + 255) / 256, 256, 0, stream>>>(fc2_w, wfc2, 1536 * 384);
  build_bias<<<(12 * NTOK * NTOK + 255) / 256, 256, 0, stream>>>(bias_table, biasP);
  build_reg<<<(64 * NTOK + 255) / 256, 256, 0, stream>>>(regT);

  // 2) LN1 + roll + window partition + gt concat
  ln1_gather<<<MROWS, 128, 0, stream>>>(x, gt, norm1_g, norm1_b, xc);

  // 3) QKV GEMM  [22016 x 384] x [384 x 1152]
  gemm_f16<0, MROWS, 384, 1152><<<dim3(172, 9), 256, 0, stream>>>(xc, wqkv,
      qkv_b, nullptr, qh, kh, vh, nullptr, nullptr, nullptr);

  // 4) windowed attention
  attn_win<<<dim3(64, 12, 6), 256, SMEM_ATTN, stream>>>(qh, kh, vh, biasP, regT, attnbuf);

  // 5) proj GEMM + unwindow + roll-back + residual (+ gt_out)
  gemm_f16<1, MROWS, 384, 384><<<dim3(172, 3), 256, 0, stream>>>(attnbuf, wproj,
      proj_b, out, nullptr, nullptr, nullptr, x, xres, nullptr);

  // 6) LN2
  ln2_kernel<<<LTOT, 128, 0, stream>>>(xres, norm2_g, norm2_b, ln2h);

  // 7) fc1 + GELU   [21952 x 384] x [384 x 1536]
  gemm_f16<2, LTOT, 384, 1536><<<dim3(172, 12), 256, 0, stream>>>(ln2h, wfc1,
      fc1_b, nullptr, nullptr, nullptr, nullptr, nullptr, nullptr, h1);

  // 8) fc2 + residual -> d_out
  gemm_f16<3, LTOT, 1536, 384><<<dim3(172, 3), 256, 0, stream>>>(h1, wfc2,
      fc2_b, out, nullptr, nullptr, nullptr, nullptr, xres, nullptr);
}
